// Discriminator_78022375899478
// MI455X (gfx1250) — matrix operand order repaired and checked by reading, not compiled, not run on hardware
//
#include <hip/hip_runtime.h>
#include <hip/hip_bf16.h>

typedef __attribute__((ext_vector_type(16))) _Float16 v16h;
typedef __attribute__((ext_vector_type(8)))  _Float16 v8h;
typedef __attribute__((ext_vector_type(8)))  float    v8f;

#define LRELU(v) ((v) >= 0.f ? (v) : 0.2f * (v))

// ---------------------------------------------------------------------------
// conv1: (32,1,32,64,64) -> (32,64,16,32,32), stride (2,2,2), pad (0,1,1), lrelu
// memory-bound (32 MACs/output); plain VALU kernel, writes f16 for WMMA layers.
// ---------------------------------------------------------------------------
__global__ __launch_bounds__(256) void conv1_lrelu_kernel(
    const float* __restrict__ x, const float* __restrict__ w1,
    _Float16* __restrict__ h1)
{
    const long long total = 32LL * 64 * 16 * 32 * 32;
    long long i = (long long)blockIdx.x * 256 + threadIdx.x;
    if (i >= total) return;
    int ow = (int)(i % 32); long long t = i / 32;
    int oh = (int)(t % 32); t /= 32;
    int od = (int)(t % 16); t /= 16;
    int oc = (int)(t % 64); t /= 64;
    int nb = (int)t;
    const float* wv = w1 + oc * 32;
    const int id0 = od * 2, ih0 = oh * 2 - 1, iw0 = ow * 2 - 1;
    float s = 0.f;
#pragma unroll
    for (int kd = 0; kd < 2; ++kd)
#pragma unroll
        for (int kh = 0; kh < 4; ++kh)
#pragma unroll
            for (int kw = 0; kw < 4; ++kw) {
                int ih = ih0 + kh, iw = iw0 + kw;
                if ((unsigned)ih < 64u && (unsigned)iw < 64u)
                    s += x[(((long long)nb * 32 + (id0 + kd)) * 64 + ih) * 64 + iw]
                         * wv[kd * 16 + kh * 4 + kw];
            }
    s = LRELU(s);
    h1[i] = (_Float16)s;
}

// ---------------------------------------------------------------------------
// f32 -> f16 weight conversion (GEMM-B layout already K-contiguous per oc row)
// ---------------------------------------------------------------------------
__global__ __launch_bounds__(256) void f32_to_f16_kernel(
    const float* __restrict__ in, _Float16* __restrict__ out, long long n)
{
    long long i = (long long)blockIdx.x * 256 + threadIdx.x;
    if (i < n) out[i] = (_Float16)in[i];
}

// ---------------------------------------------------------------------------
// Implicit-GEMM 3D conv (stride 2, pad 1) via v_wmma_f32_16x16x32_f16.
//   GEMM: M = 32*OD*OH*OW, N = OC, K = IC*64.
//   Block = 256 threads (8 waves). Block tile: 32 M-rows x 128 N-cols.
//   Each wave: one 16-col N tile x two 16-row M tiles (2 C accumulators,
//   B fragment reused across both WMMAs).
//   A staged in LDS: cooperative im2col of 32 x 128 K-chunk, padded row
//   stride (136 halfs = 272 B) -> conflict-free ds_load_b128 fragment reads.
//   A frag element e -> K = half*16 + e  (contiguous order, same as B)
//   B frag element e -> K = half*16 + e  (contiguous 32 B load) (same order as A)
// ---------------------------------------------------------------------------
template <int IC, int OC, int ID, int IH, int IW, int OD, int OH, int OW>
__global__ __launch_bounds__(256) void conv_wmma_kernel(
    const _Float16* __restrict__ A, const _Float16* __restrict__ Wf,
    float* __restrict__ Cout)
{
    constexpr int K  = IC * 64;
    constexpr int KC = 128;       // K chunk staged per barrier round
    constexpr int LDW = KC + 8;   // padded LDS row stride in halfs
    __shared__ _Float16 als[32 * LDW];

    const int tid  = threadIdx.x;
    const int lane = tid & 31;
    const int wave = tid >> 5;
    const int half = lane >> 4;
    const int mn   = lane & 15;
    const int m0   = blockIdx.x * 32;
    const int n0   = (blockIdx.y * 8 + wave) * 16;

    // fill assignment: thread -> rows (tid>>4) and (tid>>4)+16, col chunk of 8
    const int frow = tid >> 4;
    const int fcol = (tid & 15) * 8;

    const _Float16* wrow = Wf + (long long)(n0 + mn) * K;

    v8f acc0 = {}, acc1 = {};
    for (int k0 = 0; k0 < K; k0 += KC) {
        __syncthreads();
        // ---- cooperative im2col fill: 32 rows x KC halfs ----
#pragma unroll
        for (int rr = 0; rr < 2; ++rr) {
            const int row = frow + rr * 16;
            const int m  = m0 + row;
            const int ow = m % OW;
            const int oh = (m / OW) % OH;
            const int od = (m / (OW * OH)) % OD;
            const int nb = m / (OW * OH * OD);
            const int kbase = k0 + fcol;      // 8-aligned: ic, kd constant
            const int ic = kbase >> 6;
            const int r0 = kbase & 63;
            const int id = od * 2 - 1 + (r0 >> 4);
            const bool dok = (unsigned)id < (unsigned)ID;
            const long long base = (((long long)nb * IC + ic) * ID + id) * (long long)(IH * IW);
            v8h tmp;
#pragma unroll
            for (int u = 0; u < 8; ++u) {
                const int r  = r0 + u;
                const int ih = oh * 2 - 1 + ((r >> 2) & 3);
                const int iw = ow * 2 - 1 + (r & 3);
                _Float16 v = (_Float16)0.f;
                if (dok && (unsigned)ih < (unsigned)IH && (unsigned)iw < (unsigned)IW)
                    v = A[base + (long long)ih * IW + iw];
                tmp[u] = v;
            }
            *(v8h*)(&als[row * LDW + fcol]) = tmp;
        }
        __syncthreads();

        if (k0 + KC < K)  // prefetch next B chunk -> global_prefetch_b8
            __builtin_prefetch(wrow + k0 + KC + half * 16, 0, 1);

        // ---- 4 x (1 B load, 2 A ds loads each tile, 2 WMMAs) ----
#pragma unroll
        for (int kc = 0; kc < KC; kc += 32) {
            const v16h b = *(const v16h*)(wrow + k0 + kc + half * 16);
            const v8h lo0 = *(const v8h*)(&als[mn * LDW + kc + half * 16]);
            const v8h hi0 = *(const v8h*)(&als[mn * LDW + kc + 8 + half * 16]);
            const v8h lo1 = *(const v8h*)(&als[(16 + mn) * LDW + kc + half * 16]);
            const v8h hi1 = *(const v8h*)(&als[(16 + mn) * LDW + kc + 8 + half * 16]);
            const v16h a0 = __builtin_shufflevector(lo0, hi0,
                0, 1, 2, 3, 4, 5, 6, 7, 8, 9, 10, 11, 12, 13, 14, 15);
            const v16h a1 = __builtin_shufflevector(lo1, hi1,
                0, 1, 2, 3, 4, 5, 6, 7, 8, 9, 10, 11, 12, 13, 14, 15);
            acc0 = __builtin_amdgcn_wmma_f32_16x16x32_f16(
                false, a0, false, b, (short)0, acc0, false, false);
            acc1 = __builtin_amdgcn_wmma_f32_16x16x32_f16(
                false, a1, false, b, (short)0, acc1, false, false);
        }
    }

    // ---- store both C tiles (VGPR r: lanes 0-15 -> M=r, 16-31 -> M=r+8) ----
#pragma unroll
    for (int t = 0; t < 2; ++t) {
        const v8f acc = t ? acc1 : acc0;
        const int mt0 = m0 + t * 16;
        const int n   = n0 + mn;
#pragma unroll
        for (int r = 0; r < 8; ++r) {
            const int mm  = mt0 + r + 8 * half;
            const int oww = mm % OW;
            const int ohh = (mm / OW) % OH;
            const int odd = (mm / (OW * OH)) % OD;
            const int nbb = mm / (OW * OH * OD);
            Cout[((((long long)nbb * OC + n) * OD + odd) * OH + ohh) * OW + oww] = acc[r];
        }
    }
}

// ---------------------------------------------------------------------------
// Per-channel batch stats (biased var, matching jnp.var) over (N, D, H, W)
// ---------------------------------------------------------------------------
template <int OC, int S>
__global__ __launch_bounds__(256) void bn_stats_kernel(
    const float* __restrict__ C, float* __restrict__ meanvar)
{
    __shared__ float s1[256], s2[256];
    const int ch = blockIdx.x;
    float a = 0.f, b = 0.f;
    for (int t = threadIdx.x; t < 32 * S; t += 256) {
        const int nb = t / S, sp = t % S;
        const float v = C[((long long)nb * OC + ch) * S + sp];
        a += v; b += v * v;
    }
    s1[threadIdx.x] = a; s2[threadIdx.x] = b;
    __syncthreads();
    for (int st = 128; st > 0; st >>= 1) {
        if (threadIdx.x < st) {
            s1[threadIdx.x] += s1[threadIdx.x + st];
            s2[threadIdx.x] += s2[threadIdx.x + st];
        }
        __syncthreads();
    }
    if (threadIdx.x == 0) {
        const float cnt = 32.f * S;
        const float mean = s1[0] / cnt;
        meanvar[ch] = mean;
        meanvar[OC + ch] = s2[0] / cnt - mean * mean;
    }
}

// ---------------------------------------------------------------------------
// BN (batch stats) + lrelu; f16 out for next WMMA layer, f32 out for layer 4
// ---------------------------------------------------------------------------
template <int OC, int S, bool F32OUT>
__global__ __launch_bounds__(256) void bn_lrelu_kernel(
    const float* __restrict__ C, const float* __restrict__ meanvar,
    const float* __restrict__ g, const float* __restrict__ bta,
    _Float16* __restrict__ h16, float* __restrict__ h32)
{
    const long long total = 32LL * OC * S;
    long long i = (long long)blockIdx.x * 256 + threadIdx.x;
    if (i >= total) return;
    const int ch = (int)((i / S) % OC);
    const float mean = meanvar[ch], var = meanvar[OC + ch];
    float v = (C[i] - mean) * rsqrtf(var + 1e-5f) * g[ch] + bta[ch];
    v = LRELU(v);
    if (F32OUT) h32[i] = v;
    else        h16[i] = (_Float16)v;
}

// ---------------------------------------------------------------------------
// Minibatch discrimination: d(i,j) = mean |h_i - h_j| over 16384 features
// ---------------------------------------------------------------------------
__global__ __launch_bounds__(256) void mb_dist_kernel(
    const float* __restrict__ h4, float* __restrict__ d)
{
    __shared__ float sm[256];
    const int i = blockIdx.x, j = blockIdx.y;
    const float* hi = h4 + (long long)i * 16384;
    const float* hj = h4 + (long long)j * 16384;
    float a = 0.f;
    for (int t = threadIdx.x; t < 16384; t += 256) a += fabsf(hi[t] - hj[t]);
    sm[threadIdx.x] = a;
    __syncthreads();
    for (int st = 128; st > 0; st >>= 1) {
        if (threadIdx.x < st) sm[threadIdx.x] += sm[threadIdx.x + st];
        __syncthreads();
    }
    if (threadIdx.x == 0)
        d[i * 32 + j] = sm[0] / 16384.f + (i == j ? 1e6f : 0.f);
}

__global__ void mb_min_kernel(const float* __restrict__ d, float* __restrict__ s)
{
    const int i = threadIdx.x;
    if (i < 32) {
        float m = d[i * 32];
        for (int j = 1; j < 32; ++j) m = fminf(m, d[i * 32 + j]);
        s[i] = m;
    }
}

// ---------------------------------------------------------------------------
// conv5: (32,513,2,4,4) . (1,513,2,4,4) -> (32,)   (channel 512 == s[nb])
// ---------------------------------------------------------------------------
__global__ __launch_bounds__(256) void conv5_kernel(
    const float* __restrict__ h4, const float* __restrict__ s,
    const float* __restrict__ w5, float* __restrict__ out)
{
    __shared__ float sm[256];
    const int nb = blockIdx.x;
    float a = 0.f;
    for (int t = threadIdx.x; t < 513 * 32; t += 256) {
        const int c = t >> 5;
        const float hv = (c < 512) ? h4[(long long)nb * 16384 + t] : s[nb];
        a += hv * w5[t];
    }
    sm[threadIdx.x] = a;
    __syncthreads();
    for (int st = 128; st > 0; st >>= 1) {
        if (threadIdx.x < st) sm[threadIdx.x] += sm[threadIdx.x + st];
        __syncthreads();
    }
    if (threadIdx.x == 0) out[nb] = sm[0];
}

// ---------------------------------------------------------------------------
extern "C" void kernel_launch(void* const* d_in, const int* in_sizes, int n_in,
                              void* d_out, int out_size, void* d_ws, size_t ws_size,
                              hipStream_t stream)
{
    const float* x  = (const float*)d_in[0];
    const float* w1 = (const float*)d_in[1];
    const float* w2 = (const float*)d_in[2];
    const float* g2 = (const float*)d_in[3];
    const float* b2 = (const float*)d_in[4];
    const float* w3 = (const float*)d_in[5];
    const float* g3 = (const float*)d_in[6];
    const float* b3 = (const float*)d_in[7];
    const float* w4 = (const float*)d_in[8];
    const float* g4 = (const float*)d_in[9];
    const float* b4 = (const float*)d_in[10];
    const float* w5 = (const float*)d_in[11];
    float* out = (float*)d_out;

    // Workspace carve-up (region A reused: h1 -> h2 -> h3; region B: c2 -> c3 -> c4)
    char* ws = (char*)d_ws;
    size_t off = 0;
    auto carve = [&](size_t bytes) -> void* {
        void* p = ws + off;
        off += (bytes + 255) & ~(size_t)255;
        return p;
    };
    _Float16* regA = (_Float16*)carve(67108864);   // 33.5M halfs: h1 / h2 / h3
    float*    regB = (float*)   carve(33554432);   // 8.4M f32:   c2 / c3 / c4
    _Float16* wf2  = (_Float16*)carve(524288ull * 2);
    _Float16* wf3  = (_Float16*)carve(2097152ull * 2);
    _Float16* wf4  = (_Float16*)carve(8388608ull * 2);
    float*    mv2  = (float*)carve(2 * 128 * 4);
    float*    mv3  = (float*)carve(2 * 256 * 4);
    float*    mv4  = (float*)carve(2 * 512 * 4);
    float*    h4f  = (float*)carve(524288ull * 4); // post-BN layer-4 activations
    float*    dmat = (float*)carve(1024 * 4);
    float*    svec = (float*)carve(32 * 4);
    (void)ws_size; (void)in_sizes; (void)n_in; (void)out_size;

    // Layer 1 (memory-bound) + weight precision conversion
    {
        const long long n1 = 32LL * 64 * 16 * 32 * 32;
        conv1_lrelu_kernel<<<(unsigned)((n1 + 255) / 256), 256, 0, stream>>>(x, w1, regA);
        f32_to_f16_kernel<<<(524288 + 255) / 256, 256, 0, stream>>>(w2, wf2, 524288LL);
        f32_to_f16_kernel<<<(2097152 + 255) / 256, 256, 0, stream>>>(w3, wf3, 2097152LL);
        f32_to_f16_kernel<<<(8388608 + 255) / 256, 256, 0, stream>>>(w4, wf4, 8388608LL);
    }

    // Layer 2: 65536 x 128 x 4096 GEMM -> BN -> lrelu
    conv_wmma_kernel<64, 128, 16, 32, 32, 8, 16, 16>
        <<<dim3(65536 / 32, 1), 256, 0, stream>>>(regA, wf2, regB);
    bn_stats_kernel<128, 2048><<<128, 256, 0, stream>>>(regB, mv2);
    bn_lrelu_kernel<128, 2048, false>
        <<<(unsigned)((32LL * 128 * 2048 + 255) / 256), 256, 0, stream>>>(
            regB, mv2, g2, b2, regA, nullptr);

    // Layer 3: 8192 x 256 x 8192 GEMM -> BN -> lrelu
    conv_wmma_kernel<128, 256, 8, 16, 16, 4, 8, 8>
        <<<dim3(8192 / 32, 2), 256, 0, stream>>>(regA, wf3, regB);
    bn_stats_kernel<256, 256><<<256, 256, 0, stream>>>(regB, mv3);
    bn_lrelu_kernel<256, 256, false>
        <<<(unsigned)((32LL * 256 * 256 + 255) / 256), 256, 0, stream>>>(
            regB, mv3, g3, b3, regA, nullptr);

    // Layer 4: 1024 x 512 x 16384 GEMM -> BN -> lrelu (f32 out for tail)
    conv_wmma_kernel<256, 512, 4, 8, 8, 2, 4, 4>
        <<<dim3(1024 / 32, 4), 256, 0, stream>>>(regA, wf4, regB);
    bn_stats_kernel<512, 32><<<512, 256, 0, stream>>>(regB, mv4);
    bn_lrelu_kernel<512, 32, true>
        <<<(unsigned)((32LL * 512 * 32 + 255) / 256), 256, 0, stream>>>(
            regB, mv4, g4, b4, nullptr, h4f);

    // Minibatch discrimination + final conv
    mb_dist_kernel<<<dim3(32, 32), 256, 0, stream>>>(h4f, dmat);
    mb_min_kernel<<<1, 32, 0, stream>>>(dmat, svec);
    conv5_kernel<<<32, 256, 0, stream>>>(h4f, svec, w5, out);
}
